// GNN_graphpred_80779744903180
// MI455X (gfx1250) — matrix operand order repaired and checked by reading, not compiled, not run on hardware
//
#include <hip/hip_runtime.h>
#include <hip/hip_bf16.h>
#include <stdint.h>

typedef __attribute__((ext_vector_type(16))) __bf16 v16bf;
typedef __attribute__((ext_vector_type(8)))  __bf16 v8bf;
typedef __attribute__((ext_vector_type(8)))  float  v8f;
typedef unsigned int u32x4 __attribute__((ext_vector_type(4)));
typedef int          i32x4 __attribute__((ext_vector_type(4)));
typedef int          i32x8 __attribute__((ext_vector_type(8)));

#define D_    300   // emb_dim
#define HP    320   // padded D (K of gemm1, row stride of h/agg)
#define H2P   640   // padded 2D (N of gemm1, K of gemm2, row stride of hid)
#define OP_   384   // padded D out of gemm2 (row stride of h2)
#define NLAY  5
#define MAXLEN 50
#define BN_EPS 1e-5f

#ifndef __has_builtin
#define __has_builtin(x) 0
#endif
#if __has_builtin(__builtin_amdgcn_tensor_load_to_lds) && \
    __has_builtin(__builtin_amdgcn_s_wait_tensorcnt)
#define USE_TDM 1
#else
#define USE_TDM 0
#endif

// native f32 -> bf16 (RNE) via hardware cvt
static __device__ __forceinline__ unsigned short bfbits(float f) {
  union { __bf16 h; unsigned short u; } x;
  x.h = (__bf16)f;
  return x.u;
}

// ---------------- input embedding: h[i] = x_emb1[atom] + x_emb2[chir] ----------------
__global__ void embed_kernel(const int* __restrict__ x, const float* __restrict__ xe1,
                             const float* __restrict__ xe2, float* __restrict__ h, int n) {
  int i = blockIdx.x;
  if (i >= n) return;
  int a = x[2 * i], c2 = x[2 * i + 1];
  float* hr = h + (size_t)i * HP;
  for (int c = threadIdx.x; c < HP; c += blockDim.x)
    hr[c] = (c < D_) ? (xe1[a * D_ + c] + xe2[c2 * D_ + c]) : 0.f;
}

// ---------------- weight prep: transpose + pad to bf16 [n][k] ----------------
__global__ void prep_w1(const float* __restrict__ W, unsigned short* __restrict__ Wt, int total) {
  int idx = blockIdx.x * blockDim.x + threadIdx.x;
  if (idx >= total) return;
  int l = idx / (H2P * HP);
  int r = idx - l * (H2P * HP);
  int nn = r / HP;
  int k = r - nn * HP;
  float v = 0.f;
  if (nn < 2 * D_ && k < D_) v = W[(size_t)l * D_ * (2 * D_) + (size_t)k * (2 * D_) + nn];
  Wt[idx] = bfbits(v);
}
__global__ void prep_w2(const float* __restrict__ W, unsigned short* __restrict__ Wt, int total) {
  int idx = blockIdx.x * blockDim.x + threadIdx.x;
  if (idx >= total) return;
  int l = idx / (OP_ * H2P);
  int r = idx - l * (OP_ * H2P);
  int nn = r / H2P;
  int k = r - nn * H2P;
  float v = 0.f;
  if (nn < D_ && k < 2 * D_) v = W[(size_t)l * (2 * D_) * D_ + (size_t)k * D_ + nn];
  Wt[idx] = bfbits(v);
}
__global__ void prep_bias(const float* __restrict__ b1, const float* __restrict__ b2,
                          float* __restrict__ b1p, float* __restrict__ b2p) {
  int idx = blockIdx.x * blockDim.x + threadIdx.x;
  if (idx < NLAY * H2P) {
    int l = idx / H2P, c = idx % H2P;
    b1p[idx] = (c < 2 * D_) ? b1[l * (2 * D_) + c] : 0.f;
  }
  if (idx < NLAY * OP_) {
    int l = idx / OP_, c = idx % OP_;
    b2p[idx] = (c < D_) ? b2[l * D_ + c] : 0.f;
  }
}

// ---------------- message passing ----------------
__global__ void agg_init(const float* __restrict__ h, const float* __restrict__ e1,
                         const float* __restrict__ e2, float* __restrict__ agg, int n) {
  int i = blockIdx.x;
  if (i >= n) return;
  const float* hs = h + (size_t)i * HP;
  const float* p1 = e1 + 4 * D_;  // self-loop bond type 4
  const float* p2 = e2;           // direction 0
  float* ag = agg + (size_t)i * HP;
  for (int c = threadIdx.x; c < HP; c += blockDim.x)
    ag[c] = (c < D_) ? (hs[c] + p1[c] + p2[c]) : 0.f;
}
__global__ void edge_scatter(const float* __restrict__ h, const int* __restrict__ ei,
                             const int* __restrict__ ea, const float* __restrict__ e1,
                             const float* __restrict__ e2, float* __restrict__ agg, int E) {
  int e = blockIdx.x;
  if (e >= E) return;
  int src = ei[e], dst = ei[E + e];
  int b0 = ea[2 * e], b1 = ea[2 * e + 1];
  const float* hs = h + (size_t)src * HP;
  const float* p1 = e1 + b0 * D_;
  const float* p2 = e2 + b1 * D_;
  float* ag = agg + (size_t)dst * HP;
  for (int c = threadIdx.x; c < D_; c += blockDim.x)
    atomicAdd(&ag[c], hs[c] + p1[c] + p2[c]);
}

// ---------------- WMMA GEMM (generic, sync staging) ----------------
// C[M x 128cols(strip)] = A[M x KP] @ Bt^T + bias, Bt pre-transposed [ncol][KP] bf16.
// Block: 256 thr = 8 waves as 4 rowgroups x 2 colgroups. Tile: 64 rows x 128 cols.
template <bool AF32, bool OUTBF16RELU, int KP, int LDOUT>
__global__ void gemm_wmma(const void* __restrict__ Ap, const unsigned short* __restrict__ Bt,
                          const float* __restrict__ bias, void* __restrict__ Out, int M) {
  extern __shared__ char smem[];
  constexpr int LDA = 72;        // sA row stride (ushort): 64 K + 8 pad
  constexpr int LDB = KP + 8;    // sB col stride (ushort)
  unsigned short* sA = (unsigned short*)smem;                  // 64 x 72
  unsigned short* sB = (unsigned short*)(smem + 64 * LDA * 2); // 128 x LDB
  const int tid = threadIdx.x;
  const int rowBase = blockIdx.x * 64;
  const int colBase = blockIdx.y * 128;

  {  // stage resident B strip: 128 cols x KP (contiguous per column)
    int col = tid >> 1, sub = tid & 1;
    constexpr int kchunk = KP >> 1;
    const unsigned short* g = Bt + (size_t)(colBase + col) * KP + sub * kchunk;
    unsigned short* s = sB + col * LDB + sub * kchunk;
    for (int k = 0; k < kchunk; k += 8)
      *(int4*)(s + k) = *(const int4*)(g + k);
  }
  __syncthreads();

  const int lane = tid & 31;
  const int w = tid >> 5;
  const int rg = w & 3;
  const int cg = w >> 2;
  const int m = lane & 15;
  const int kh = lane >> 4;

  v8f acc[4];
#pragma unroll
  for (int j = 0; j < 4; ++j)
#pragma unroll
    for (int i = 0; i < 8; ++i) acc[j][i] = 0.f;

  constexpr int ksteps = KP >> 6;
  const int arow = tid >> 2;
  const int asub = tid & 3;

  const unsigned short* aF = sA + (rg * 16 + m) * LDA + kh * 16;
  const unsigned short* bF = sB + (cg * 64 + m) * LDB + kh * 16;

  for (int ks = 0; ks < ksteps; ++ks) {
    {  // stage A tile: 64 rows x 64 K
      int gr = rowBase + arow;
      unsigned short* s = sA + arow * LDA + asub * 16;
      if (AF32) {
        const float* g = (const float*)Ap + (size_t)gr * KP + ks * 64 + asub * 16;
        unsigned int words[8];
        if (gr < M) {
#pragma unroll
          for (int i = 0; i < 8; ++i)
            words[i] = (unsigned)bfbits(g[2 * i]) | ((unsigned)bfbits(g[2 * i + 1]) << 16);
        } else {
#pragma unroll
          for (int i = 0; i < 8; ++i) words[i] = 0u;
        }
        ((int4*)s)[0] = make_int4(words[0], words[1], words[2], words[3]);
        ((int4*)(s + 8))[0] = make_int4(words[4], words[5], words[6], words[7]);
      } else {
        const unsigned short* g =
            (const unsigned short*)Ap + (size_t)gr * KP + ks * 64 + asub * 16;
        if (gr < M) {
          ((int4*)s)[0] = *(const int4*)(g);
          ((int4*)(s + 8))[0] = *(const int4*)(g + 8);
        } else {
          ((int4*)s)[0] = make_int4(0, 0, 0, 0);
          ((int4*)(s + 8))[0] = make_int4(0, 0, 0, 0);
        }
      }
    }
    __syncthreads();

#pragma unroll
    for (int kk = 0; kk < 2; ++kk) {
      const unsigned short* ap = aF + kk * 32;
      v8bf alo = *(const v8bf*)(ap);
      v8bf ahi = *(const v8bf*)(ap + 8);
      v16bf afrag = __builtin_shufflevector(alo, ahi, 0, 1, 2, 3, 4, 5, 6, 7,
                                            8, 9, 10, 11, 12, 13, 14, 15);
      v16bf bfrag[4];
#pragma unroll
      for (int j = 0; j < 4; ++j) {
        const unsigned short* bp = bF + j * 16 * LDB + ks * 64 + kk * 32;
        v8bf blo = *(const v8bf*)(bp);
        v8bf bhi = *(const v8bf*)(bp + 8);
        bfrag[j] = __builtin_shufflevector(blo, bhi, 0, 1, 2, 3, 4, 5, 6, 7,
                                           8, 9, 10, 11, 12, 13, 14, 15);
      }
#pragma unroll
      for (int j = 0; j < 4; ++j)
        acc[j] = __builtin_amdgcn_wmma_f32_16x16x32_bf16(
            false, afrag, false, bfrag[j], (short)0, acc[j], false, false);
    }
    __syncthreads();
  }

#pragma unroll
  for (int j = 0; j < 4; ++j) {
    int col = colBase + cg * 64 + j * 16 + m;
    float bv = bias[col];
#pragma unroll
    for (int i = 0; i < 8; ++i) {
      int gr = rowBase + rg * 16 + kh * 8 + i;
      if (gr < M) {
        float v = acc[j][i] + bv;
        if (OUTBF16RELU) {
          v = v > 0.f ? v : 0.f;
          ((unsigned short*)Out)[(size_t)gr * LDOUT + col] = bfbits(v);
        } else {
          ((float*)Out)[(size_t)gr * LDOUT + col] = v;
        }
      }
    }
  }
}

#if USE_TDM
// ---- TDM descriptor issue: 2D tile 64 rows x 64 k (bf16), row stride KP elements,
// ---- LDS padded 4 DWORDs after every 32 DWORDs (-> row stride 72 ushorts).
static __device__ __forceinline__ void tdm_load_tile(const unsigned short* gsrc,
                                                     unsigned lds_addr) {
  unsigned long long ga = (unsigned long long)(uintptr_t)gsrc;
  u32x4 g0 = {1u,  // count=1, user mode
              lds_addr,
              (unsigned)(ga & 0xFFFFFFFFu),
              (unsigned)((ga >> 32) & 0x1FFFFFFu) | (2u << 30)};  // type=2 (image)
  i32x8 g1 = {(int)((1u << 16) | (1u << 20) | (4u << 22) | (3u << 25)),
              // data_size=2B, pad_enable, pad_interval=32dw, pad_amount=4dw
              (int)(64u << 16),  // tensor_dim0 = 64 (bits 79:48)
              (int)(64u << 16),  // tensor_dim1 = 64 (bits 111:80)
              (int)(64u << 16),  // tile_dim0 = 64 (bits 127:112)
              64,                // tile_dim1 = 64, tile_dim2 = 0
              H2P,               // tensor_dim0_stride = 640 elements
              0, 0};
  i32x4 z4 = {0, 0, 0, 0};
#if __clang_major__ >= 23
  i32x8 z8 = {0, 0, 0, 0, 0, 0, 0, 0};
  __builtin_amdgcn_tensor_load_to_lds(g0, g1, z4, z4, z8, 0);
#else
  __builtin_amdgcn_tensor_load_to_lds(g0, g1, z4, z4, 0);
#endif
}

// ---- GEMM2 with TDM double-buffered A staging: h2 = hid @ W2t^T + b2 ----
__global__ void gemm2_wmma_tdm(const unsigned short* __restrict__ Ap,
                               const unsigned short* __restrict__ Bt,
                               const float* __restrict__ bias,
                               float* __restrict__ Out, int M) {
  constexpr int KP = H2P;     // 640
  constexpr int LDOUT = OP_;  // 384
  constexpr int LDA = 72;
  constexpr int LDB = KP + 8;
  extern __shared__ char smem[];
  unsigned short* sA = (unsigned short*)smem;                      // 2 x (64 x 72)
  unsigned short* sB = (unsigned short*)(smem + 2 * 64 * LDA * 2);
  const int tid = threadIdx.x;
  const int rowBase = blockIdx.x * 64;
  const int colBase = blockIdx.y * 128;
  const int w = tid >> 5;

  // kick TDM for stage 0 (wave 0 only; wave-level DMA, tracked by TENSORcnt)
  if (w == 0)
    tdm_load_tile(Ap + (size_t)rowBase * KP, (unsigned)(uintptr_t)sA);

  {  // stage resident B strip while TDM runs
    int col = tid >> 1, sub = tid & 1;
    constexpr int kchunk = KP >> 1;
    const unsigned short* g = Bt + (size_t)(colBase + col) * KP + sub * kchunk;
    unsigned short* s = sB + col * LDB + sub * kchunk;
    for (int k = 0; k < kchunk; k += 8)
      *(int4*)(s + k) = *(const int4*)(g + k);
  }
  __builtin_amdgcn_s_wait_tensorcnt(0);  // no-op on waves with TENSORcnt==0
  __syncthreads();

  const int lane = tid & 31;
  const int rg = w & 3;
  const int cg = w >> 2;
  const int m = lane & 15;
  const int kh = lane >> 4;

  v8f acc[4];
#pragma unroll
  for (int j = 0; j < 4; ++j)
#pragma unroll
    for (int i = 0; i < 8; ++i) acc[j][i] = 0.f;

  constexpr int ksteps = KP >> 6;  // 10
  const unsigned short* bF = sB + (cg * 64 + m) * LDB + kh * 16;

  for (int ks = 0; ks < ksteps; ++ks) {
    // issue DMA for next stage into the other buffer; overlaps with 8 WMMAs below
    if (w == 0 && ks + 1 < ksteps)
      tdm_load_tile(Ap + (size_t)rowBase * KP + (ks + 1) * 64,
                    (unsigned)(uintptr_t)(sA + ((ks + 1) & 1) * (64 * LDA)));

    const unsigned short* aF =
        sA + (ks & 1) * (64 * LDA) + (rg * 16 + m) * LDA + kh * 16;
#pragma unroll
    for (int kk = 0; kk < 2; ++kk) {
      const unsigned short* ap = aF + kk * 32;
      v8bf alo = *(const v8bf*)(ap);
      v8bf ahi = *(const v8bf*)(ap + 8);
      v16bf afrag = __builtin_shufflevector(alo, ahi, 0, 1, 2, 3, 4, 5, 6, 7,
                                            8, 9, 10, 11, 12, 13, 14, 15);
      v16bf bfrag[4];
#pragma unroll
      for (int j = 0; j < 4; ++j) {
        const unsigned short* bp = bF + j * 16 * LDB + ks * 64 + kk * 32;
        v8bf blo = *(const v8bf*)(bp);
        v8bf bhi = *(const v8bf*)(bp + 8);
        bfrag[j] = __builtin_shufflevector(blo, bhi, 0, 1, 2, 3, 4, 5, 6, 7,
                                           8, 9, 10, 11, 12, 13, 14, 15);
      }
#pragma unroll
      for (int j = 0; j < 4; ++j)
        acc[j] = __builtin_amdgcn_wmma_f32_16x16x32_bf16(
            false, afrag, false, bfrag[j], (short)0, acc[j], false, false);
    }
    __builtin_amdgcn_s_wait_tensorcnt(0);
    __syncthreads();
  }

#pragma unroll
  for (int j = 0; j < 4; ++j) {
    int col = colBase + cg * 64 + j * 16 + m;
    float bv = bias[col];
#pragma unroll
    for (int i = 0; i < 8; ++i) {
      int gr = rowBase + rg * 16 + kh * 8 + i;
      if (gr < M) Out[(size_t)gr * LDOUT + col] = acc[j][i] + bv;
    }
  }
}
#endif  // USE_TDM

// ---------------- batch norm ----------------
__global__ void zero_stats(float* s) { s[threadIdx.x] = 0.f; }  // 768 threads

__global__ void bn_partial(const float* __restrict__ h2, float* __restrict__ stats,
                           int n, int rowsPer) {
  int c0 = threadIdx.x;
  int c1 = 256 + threadIdx.x;
  int r0 = blockIdx.x * rowsPer;
  int r1 = r0 + rowsPer;
  if (r1 > n) r1 = n;
  float s0 = 0.f, q0 = 0.f, s1 = 0.f, q1 = 0.f;
  for (int r = r0; r < r1; ++r) {
    float v = h2[(size_t)r * OP_ + c0];
    s0 += v; q0 += v * v;
    if (c1 < D_) {
      float u = h2[(size_t)r * OP_ + c1];
      s1 += u; q1 += u * u;
    }
  }
  atomicAdd(&stats[c0], s0);
  atomicAdd(&stats[OP_ + c0], q0);
  if (c1 < D_) {
    atomicAdd(&stats[c1], s1);
    atomicAdd(&stats[OP_ + c1], q1);
  }
}
__global__ void bn_finalize(const float* __restrict__ stats, float* __restrict__ mv, int n) {
  int c = threadIdx.x;
  float inv_n = 1.f / (float)n;
  float mean = stats[c] * inv_n;
  float var = stats[OP_ + c] * inv_n - mean * mean;
  mv[c] = mean;
  mv[OP_ + c] = rsqrtf(var + BN_EPS);
}
__global__ void bn_apply(const float* __restrict__ h2, const float* __restrict__ mv,
                         const float* __restrict__ gamma, const float* __restrict__ beta,
                         float* __restrict__ h, int n, int do_relu) {
  int i = blockIdx.x;
  if (i >= n) return;
  const float* row = h2 + (size_t)i * OP_;
  float* out = h + (size_t)i * HP;
  for (int c = threadIdx.x; c < D_; c += blockDim.x) {
    float v = (row[c] - mv[c]) * mv[OP_ + c] * gamma[c] + beta[c];
    if (do_relu) v = v > 0.f ? v : 0.f;
    out[c] = v;
  }
}

// ---------------- pooling / padding ----------------
__global__ void bounds_init(int* gs, int* ge, int B, int n) {
  int g = blockIdx.x * blockDim.x + threadIdx.x;
  if (g < B) { gs[g] = n; ge[g] = 0; }
}
__global__ void bounds_fill(const int* __restrict__ batch, int* gs, int* ge, int n) {
  int i = blockIdx.x * blockDim.x + threadIdx.x;
  if (i < n) {
    int b = batch[i];
    atomicMin(&gs[b], i);
    atomicMax(&ge[b], i + 1);
  }
}
__global__ void pool_mean(const float* __restrict__ h, const int* __restrict__ gs,
                          const int* __restrict__ ge, float* __restrict__ out) {
  int g = blockIdx.x;
  int s = gs[g], e = ge[g];
  int cnt = (e > s) ? (e - s) : 0;
  float inv = 1.f / (float)(cnt > 1 ? cnt : 1);
  for (int c = threadIdx.x; c < D_; c += blockDim.x) {
    float acc = 0.f;
    for (int r = s; r < e; ++r) acc += h[(size_t)r * HP + c];
    out[(size_t)g * D_ + c] = acc * inv;
  }
}
__global__ void fzero(float* __restrict__ p, size_t n) {
  size_t i = (size_t)blockIdx.x * blockDim.x + threadIdx.x;
  size_t stride = (size_t)gridDim.x * blockDim.x;
  for (; i < n; i += stride) p[i] = 0.f;
}
__global__ void pad_nodes(const float* __restrict__ h, const int* __restrict__ batch,
                          const int* __restrict__ gs, float* __restrict__ nf,
                          float* __restrict__ mask, int n) {
  int i = blockIdx.x;
  if (i >= n) return;
  int b = batch[i];
  int pos = i - gs[b];
  if (pos >= MAXLEN) return;
  float* dst = nf + ((size_t)b * MAXLEN + pos) * D_;
  const float* src = h + (size_t)i * HP;
  for (int c = threadIdx.x; c < D_; c += blockDim.x) dst[c] = src[c];
  if (threadIdx.x == 0) mask[(size_t)b * MAXLEN + pos] = 1.f;
}

// ---------------- launch ----------------
extern "C" void kernel_launch(void* const* d_in, const int* in_sizes, int n_in,
                              void* d_out, int out_size, void* d_ws, size_t ws_size,
                              hipStream_t stream) {
  (void)n_in; (void)ws_size;
  const int* x     = (const int*)d_in[0];
  const int* ei    = (const int*)d_in[1];
  const int* ea    = (const int*)d_in[2];
  const int* batch = (const int*)d_in[3];
  const float* xe1 = (const float*)d_in[5];
  const float* xe2 = (const float*)d_in[6];
  const float* ee1 = (const float*)d_in[7];
  const float* ee2 = (const float*)d_in[8];
  const float* W1  = (const float*)d_in[9];
  const float* b1  = (const float*)d_in[10];
  const float* W2  = (const float*)d_in[11];
  const float* b2  = (const float*)d_in[12];
  const float* gam = (const float*)d_in[13];
  const float* bet = (const float*)d_in[14];

  const int n = in_sizes[0] / 2;
  const int E = in_sizes[1] / 2;
  const int B = out_size / (D_ + MAXLEN * D_ + MAXLEN);

  char* ws = (char*)d_ws;
  size_t off = 0;
  auto carve = [&](size_t bytes) -> void* {
    void* p = ws + off;
    off = (off + bytes + 255) & ~(size_t)255;
    return p;
  };
  float* h            = (float*)carve((size_t)n * HP * sizeof(float));
  float* agg          = (float*)carve((size_t)n * HP * sizeof(float));
  unsigned short* hid = (unsigned short*)carve((size_t)n * H2P * sizeof(unsigned short));
  float* h2           = (float*)carve((size_t)n * OP_ * sizeof(float));
  unsigned short* W1t = (unsigned short*)carve((size_t)NLAY * H2P * HP * 2);
  unsigned short* W2t = (unsigned short*)carve((size_t)NLAY * OP_ * H2P * 2);
  float* b1p          = (float*)carve((size_t)NLAY * H2P * sizeof(float));
  float* b2p          = (float*)carve((size_t)NLAY * OP_ * sizeof(float));
  float* stats        = (float*)carve(2 * OP_ * sizeof(float));
  float* mv           = (float*)carve(2 * OP_ * sizeof(float));
  int* gs             = (int*)carve((size_t)B * sizeof(int));
  int* ge             = (int*)carve((size_t)B * sizeof(int));

  embed_kernel<<<n, 128, 0, stream>>>(x, xe1, xe2, h, n);
  int tw1 = NLAY * H2P * HP;
  prep_w1<<<(tw1 + 255) / 256, 256, 0, stream>>>(W1, W1t, tw1);
  int tw2 = NLAY * OP_ * H2P;
  prep_w2<<<(tw2 + 255) / 256, 256, 0, stream>>>(W2, W2t, tw2);
  prep_bias<<<(NLAY * H2P + 255) / 256, 256, 0, stream>>>(b1, b2, b1p, b2p);

  const size_t smem1 = 64 * 72 * 2 + 128 * (size_t)(HP + 8) * 2;       // ~91 KB
  const size_t smem2 = 2 * 64 * 72 * 2 + 128 * (size_t)(H2P + 8) * 2;  // ~180 KB (dbl buf)
  const dim3 g1((n + 63) / 64, H2P / 128);  // 5 col strips
  const dim3 g2((n + 63) / 64, OP_ / 128);  // 3 col strips

  for (int l = 0; l < NLAY; ++l) {
    const float* e1l = ee1 + (size_t)l * 6 * D_;
    const float* e2l = ee2 + (size_t)l * 3 * D_;
    agg_init<<<n, 128, 0, stream>>>(h, e1l, e2l, agg, n);
    edge_scatter<<<E, 128, 0, stream>>>(h, ei, ea, e1l, e2l, agg, E);

    gemm_wmma<true, true, HP, H2P><<<g1, 256, smem1, stream>>>(
        agg, W1t + (size_t)l * H2P * HP, b1p + (size_t)l * H2P, hid, n);
#if USE_TDM
    gemm2_wmma_tdm<<<g2, 256, smem2, stream>>>(
        hid, W2t + (size_t)l * OP_ * H2P, b2p + (size_t)l * OP_, h2, n);
#else
    gemm_wmma<false, false, H2P, OP_><<<g2, 256, smem2, stream>>>(
        hid, W2t + (size_t)l * OP_ * H2P, b2p + (size_t)l * OP_, h2, n);
#endif

    zero_stats<<<1, 2 * OP_, 0, stream>>>(stats);
    int rowsPer = (n + 511) / 512;
    bn_partial<<<512, 256, 0, stream>>>(h2, stats, n, rowsPer);
    bn_finalize<<<1, OP_, 0, stream>>>(stats, mv, n);
    bn_apply<<<n, 128, 0, stream>>>(h2, mv, gam + (size_t)l * D_, bet + (size_t)l * D_,
                                    h, n, l != NLAY - 1);
  }

  float* out = (float*)d_out;
  bounds_init<<<(B + 255) / 256, 256, 0, stream>>>(gs, ge, B, n);
  bounds_fill<<<(n + 255) / 256, 256, 0, stream>>>(batch, gs, ge, n);
  pool_mean<<<B, 128, 0, stream>>>(h, gs, ge, out);

  float* nf = out + (size_t)B * D_;
  float* mask = nf + (size_t)B * MAXLEN * D_;
  size_t tail = (size_t)B * MAXLEN * D_ + (size_t)B * MAXLEN;
  fzero<<<2048, 256, 0, stream>>>(nf, tail);
  pad_nodes<<<n, 128, 0, stream>>>(h, batch, gs, nf, mask, n);
}
